// InfoNCELoss_14044543058506
// MI455X (gfx1250) — matrix operand order repaired and checked by reading, not compiled, not run on hardware
//
#include <hip/hip_runtime.h>
#include <hip/hip_bf16.h>
#include <math.h>

typedef __attribute__((ext_vector_type(16))) __bf16 v16bf;
typedef __attribute__((ext_vector_type(8)))  __bf16 v8bf;
typedef __attribute__((ext_vector_type(8)))  float  v8f;

#define DDIM   512
#define BROWS  4096
#define PROWS  8192
#define INV_T  14.2857142857142857f
#define EPSV   1e-8f

#define USE_ASYNC_STAGE 1   // global_load_async_to_lds_b128 staging
#define USE_DS_TR16     1   // ds_load_b128 B-fragment build

// ---------------------------------------------------------------------------
// Kernel 1: L2-normalize anchors -> f32 copy + bf16 copy; zero accumulators.
// ---------------------------------------------------------------------------
__global__ void __launch_bounds__(256)
norm_kernel(const float* __restrict__ A, float* __restrict__ an,
            __bf16* __restrict__ abf, float* __restrict__ negsum,
            float* __restrict__ out) {
  const int row = blockIdx.x;
  const int t = threadIdx.x;
  const float* src = A + (size_t)row * DDIM;
  float x0 = src[t];
  float x1 = src[t + 256];
  float ss = x0 * x0 + x1 * x1;
  #pragma unroll
  for (int m = 16; m >= 1; m >>= 1) ss += __shfl_xor(ss, m, 32);
  __shared__ float wsum[8];
  if ((t & 31) == 0) wsum[t >> 5] = ss;
  __syncthreads();
  float tot = 0.f;
  #pragma unroll
  for (int i = 0; i < 8; i++) tot += wsum[i];
  const float rn = rsqrtf(tot);
  const float y0 = x0 * rn, y1 = x1 * rn;
  an[(size_t)row * DDIM + t]        = y0;
  an[(size_t)row * DDIM + t + 256]  = y1;
  abf[(size_t)row * DDIM + t]       = (__bf16)y0;
  abf[(size_t)row * DDIM + t + 256] = (__bf16)y1;
  if (t == 0) negsum[row] = 0.f;
  if (row == 0 && t == 0) out[0] = 0.f;
}

// ---------------------------------------------------------------------------
// ds_load_b128 wrapper: plain 8x16-bit (b128) load from row-major LDS (no transpose).
// Per-lane address: tile_base + (lane&15)*row_stride + (lane>=16)*16B.
// ---------------------------------------------------------------------------
__device__ __forceinline__ v8bf ds_tr16_load(unsigned lds_byte_addr) {
  v8bf d;
  asm volatile("ds_load_b128 %0, %1"
               : "=v"(d) : "v"(lds_byte_addr) : "memory");
  return d;
}

// ---------------------------------------------------------------------------
// Kernel 2: fused Gram (a @ a.T) + exp + row-sum (diag excluded).
// Block = 128 threads (4 waves), tile = 64x64, K = 512.
// B panel staged row-major in LDS via async b128 copies; B fragments built
// with plain ds_load_b128; A fragments register-cached for the whole K.
// ---------------------------------------------------------------------------
__global__ void __launch_bounds__(128)
gram_kernel(const __bf16* __restrict__ abf, float* __restrict__ negsum) {
  __shared__ __align__(16) __bf16 Bsm[64 * DDIM];   // 64 KB row-major panel

  const int bi = blockIdx.y;   // row block
  const int bj = blockIdx.x;   // col block
  const int t  = threadIdx.x;
  const unsigned lds0 = (unsigned)(size_t)(&Bsm[0]);

  // ---- Stage B panel: rows bj*64 .. +63 (64 x 512 bf16 = 4096 x b128) ----
  {
    const __bf16* gp = abf + (size_t)bj * 64 * DDIM;
#if USE_ASYNC_STAGE
    #pragma unroll
    for (int i = 0; i < 32; i++) {
      const unsigned voff  = (unsigned)((t + i * 128) * 16);  // byte offset
      const unsigned laddr = lds0 + voff;
      asm volatile("global_load_async_to_lds_b128 %0, %1, %2"
                   :: "v"(laddr), "v"(voff), "s"(gp) : "memory");
    }
    asm volatile("s_wait_asynccnt 0x0" ::: "memory");
#else
    const uint4* g = (const uint4*)gp;
    uint4* s = (uint4*)Bsm;
    #pragma unroll
    for (int i = 0; i < 32; i++) s[t + i * 128] = g[t + i * 128];
#endif
  }
  __syncthreads();

  const int w    = t >> 5;
  const int l    = t & 31;
  const int half = l >> 4;          // lane half: K interleave / M offset
  const int ln   = l & 15;
  const int i0   = bi * 64 + w * 16;             // wave's 16 output rows
  const int koff = half * 8;

  // ---- A fragments for the full K=512, cached in registers (16 x v16bf) ---
  const __bf16* ap = abf + (size_t)(i0 + ln) * DDIM;
  v16bf afrag[16];
  #pragma unroll
  for (int ks = 0; ks < 16; ks++) {
    v8bf lo = *(const v8bf*)(ap + ks * 32 + koff);
    v8bf hi = *(const v8bf*)(ap + ks * 32 + 16 + koff);
    afrag[ks] = __builtin_shufflevector(lo, hi, 0, 1, 2, 3, 4, 5, 6, 7,
                                                8, 9, 10, 11, 12, 13, 14, 15);
  }

  float rs[8];
  #pragma unroll
  for (int r = 0; r < 8; r++) rs[r] = 0.f;

  for (int cb = 0; cb < 4; cb++) {
    const int j0l = cb * 16;
    v8f c = {};
    // Per-lane base address into the 16x16 tile for ds_load_b128:
    // row (j0l + ln), byte column half*16 within the tile.
    const unsigned tbase = lds0 + (unsigned)(((j0l + ln) * DDIM) * 2 + half * 16);
    for (int kk = 0; kk < DDIM; kk += 32) {
#if USE_DS_TR16
      v8bf blo = ds_tr16_load(tbase + (unsigned)(kk * 2));        // K kk..kk+15
      v8bf bhi = ds_tr16_load(tbase + (unsigned)(kk * 2) + 32);   // K kk+16..kk+31
      asm volatile("s_wait_dscnt 0x0" ::: "memory");
      v16bf bfrag = __builtin_shufflevector(blo, bhi, 0, 1, 2, 3, 4, 5, 6, 7,
                                                      8, 9, 10, 11, 12, 13, 14, 15);
#else
      v16bf bfrag;
      const int kl = kk + l;
      #pragma unroll
      for (int e = 0; e < 16; e++) bfrag[e] = Bsm[(j0l + e) * DDIM + kl];
#endif
      c = __builtin_amdgcn_wmma_f32_16x16x32_bf16(
              false, afrag[kk >> 5], false, bfrag, (short)0, c, false, false);
    }
    // Epilogue: scale 1/T, exp, exclude diagonal, accumulate row sums.
    const int col = bj * 64 + j0l + ln;
    #pragma unroll
    for (int r = 0; r < 8; r++) {
      const int row = i0 + r + 8 * half;   // C layout: lanes 16-31 hold M=r+8
      float e = (row == col) ? 0.f : __expf(c[r] * INV_T);
      rs[r] += e;
    }
  }

  // Reduce over the 16 columns (lanes within each half) and accumulate.
  #pragma unroll
  for (int r = 0; r < 8; r++) {
    float v = rs[r];
    v += __shfl_xor(v, 1, 16);
    v += __shfl_xor(v, 2, 16);
    v += __shfl_xor(v, 4, 16);
    v += __shfl_xor(v, 8, 16);
    if (ln == 0) atomicAdd(&negsum[i0 + r + 8 * half], v);
  }
}

// ---------------------------------------------------------------------------
// Kernel 3: pos similarities + loss. One wave per (b,p) pair, full f32.
// ---------------------------------------------------------------------------
__global__ void __launch_bounds__(256)
loss_kernel(const float* __restrict__ pos_in, const float* __restrict__ an,
            const float* __restrict__ negsum, float* __restrict__ out) {
  const int pair = blockIdx.x * 8 + (threadIdx.x >> 5);
  const int l = threadIdx.x & 31;
  const int b = pair >> 1;                       // reshape(B, P=2, D)
  const float* pr = pos_in + (size_t)pair * DDIM;
  const float* ar = an + (size_t)b * DDIM;
  float ss = 0.f, dt = 0.f;
  #pragma unroll
  for (int i = 0; i < 16; i++) {
    float pv = pr[l + 32 * i];
    float av = ar[l + 32 * i];
    ss += pv * pv;
    dt += av * pv;
  }
  #pragma unroll
  for (int m = 16; m >= 1; m >>= 1) {
    ss += __shfl_xor(ss, m, 32);
    dt += __shfl_xor(dt, m, 32);
  }
  if (l == 0) {
    const float pos = dt * rsqrtf(ss) * INV_T;
    const float ep  = __expf(pos);
    const float loss = logf(ep + negsum[b] + EPSV) - pos;
    atomicAdd(out, loss * (1.0f / (float)PROWS));
  }
}

// ---------------------------------------------------------------------------
extern "C" void kernel_launch(void* const* d_in, const int* in_sizes, int n_in,
                              void* d_out, int out_size, void* d_ws, size_t ws_size,
                              hipStream_t stream) {
  const float* anchor   = (const float*)d_in[0];   // [4096, 512] f32
  const float* positive = (const float*)d_in[1];   // [8192, 512] f32
  float* out = (float*)d_out;

  char* ws = (char*)d_ws;
  float*  an     = (float*)ws;                            // 8 MB  normalized anchors f32
  __bf16* abf    = (__bf16*)(ws + 8u * 1024 * 1024);      // 4 MB  normalized anchors bf16
  float*  negsum = (float*)(ws + 12u * 1024 * 1024);      // 16 KB row exp-sums (minus diag)

  norm_kernel<<<BROWS, 256, 0, stream>>>(anchor, an, abf, negsum, out);
  gram_kernel<<<dim3(64, 64), 128, 0, stream>>>(abf, negsum);
  loss_kernel<<<PROWS / 8, 256, 0, stream>>>(positive, an, negsum, out);
}
